// GaussianHead_111669149986
// MI455X (gfx1250) — matrix operand order repaired and checked by reading, not compiled, not run on hardware
//
#include <hip/hip_runtime.h>
#include <math.h>

typedef __attribute__((ext_vector_type(16))) _Float16 v16h;
typedef __attribute__((ext_vector_type(8)))  _Float16 v8h;
typedef __attribute__((ext_vector_type(8)))  float    v8f;
typedef __attribute__((ext_vector_type(4)))  int      v4i;

#define GLOBAL_AS __attribute__((address_space(1)))
#define LOCAL_AS  __attribute__((address_space(3)))

#define CCH 64
#define HH 512
#define WW 512
#define HWPIX (HH * WW)
#define TILE_W 128           // output pixels per block (4 waves x 32 px)
#define LDSP 132             // staged pixels per row (130 used, padded)
#define WP_HALVES 36864      // 9 taps * 2 kchunks * 4 ntiles * 32 lanes * 16 halves
#define CONV_THREADS 128     // 4 waves; each wave owns 2 M-tiles (32 px) x 64 ch

#if defined(__has_builtin)
#  if __has_builtin(__builtin_amdgcn_global_load_async_to_lds_b128)
#    define HAVE_ASYNC_LDS 1
#  endif
#endif
#ifndef HAVE_ASYNC_LDS
#  define HAVE_ASYNC_LDS 0
#endif

// ---------------------------------------------------------------------------
// NCHW fp32 -> NHWC fp16
// ---------------------------------------------------------------------------
__global__ __launch_bounds__(256) void pack_feat(const float* __restrict__ in,
                                                 _Float16* __restrict__ out) {
    size_t i = (size_t)blockIdx.x * 256 + threadIdx.x;   // 64*512*512 threads
    int c = (int)(i / HWPIX);
    int pix = (int)(i % HWPIX);
    out[(size_t)pix * CCH + c] = (_Float16)in[i];
}

// ---------------------------------------------------------------------------
// OIHW fp32 weights -> B-fragment-packed fp16: [tap][q][ntile][lane][16]
// B 32x16 layout: lanes 0-15 hold K=0..7,16..23 (halves 0..15), lanes 16-31 hold K=8..15,24..31
// ---------------------------------------------------------------------------
__global__ __launch_bounds__(256) void pack_w(const float* __restrict__ w,
                                              _Float16* __restrict__ wp) {
    int t = blockIdx.x * 256 + threadIdx.x;
    if (t >= WP_HALVES) return;
    int h    = t & 15;
    int lane = (t >> 4) & 31;
    int nt   = (t >> 9) & 3;
    int q    = (t >> 11) & 1;
    int tap  = t >> 12;                 // 0..8
    int co = nt * 16 + (lane & 15);     // N
    int K  = (lane >> 4) * 8 + (h & 7) + (h >> 3) * 16;      // K within 32
    int ci = q * 32 + K;
    int kh = tap / 3, kw = tap % 3;
    wp[t] = (_Float16)w[(((size_t)co * CCH + ci) * 3 + kh) * 3 + kw];
}

// ---------------------------------------------------------------------------
// 3x3 conv, stride 1, SAME, no bias.  in/out: NHWC fp16 [H][W][64].
// One block = one row segment of 128 pixels; wave w owns pixels [w*32, w*32+32)
// (two WMMA M-tiles) x all 64 output channels (4 WMMA N-tiles).
// Per k-step: 8 B global loads + 4 A LDS loads feed 8 WMMAs.
// ---------------------------------------------------------------------------
__global__ __launch_bounds__(CONV_THREADS) void conv3x3_wmma(
        const _Float16* __restrict__ in,
        const _Float16* __restrict__ wp,
        _Float16* __restrict__ out) {
    __shared__ _Float16 sm[3 * LDSP * CCH];   // 50688 bytes

    const int c0  = blockIdx.x * TILE_W;
    const int r   = blockIdx.y;
    const int tid = threadIdx.x;

    // ---- stage 3 rows x 130 px x 64 ch into LDS (zero-fill OOB halo) ----
    const int items = 3 * 130 * 8;            // 16-byte chunks
    for (int it = tid; it < items; it += CONV_THREADS) {
        int chunk = it & 7;
        int p     = (it >> 3) % 130;
        int row   = (it >> 3) / 130;
        int gr = r + row - 1;
        int gc = c0 + p - 1;
        _Float16* lp = sm + ((size_t)row * LDSP + p) * CCH + chunk * 8;
        if ((unsigned)gr < HH && (unsigned)gc < WW) {
            const _Float16* gp = in + ((size_t)gr * WW + gc) * CCH + chunk * 8;
#if HAVE_ASYNC_LDS
            __builtin_amdgcn_global_load_async_to_lds_b128(
                (GLOBAL_AS v4i*)gp, (LOCAL_AS v4i*)lp, 0, 0);
#else
            *(v8h*)lp = *(const v8h*)gp;
#endif
        } else {
            v8h z = {};
            *(v8h*)lp = z;
        }
    }
#if HAVE_ASYNC_LDS
#  if __has_builtin(__builtin_amdgcn_s_wait_asynccnt)
    __builtin_amdgcn_s_wait_asynccnt(0);
#  else
    asm volatile("s_wait_asynccnt 0x0" ::: "memory");
#  endif
#endif
    __syncthreads();

    const int wave = tid >> 5;                // 0..3
    const int lane = tid & 31;
    const int m    = lane & 15;               // A-matrix M = pixel within M-tile
    const int kb   = (lane >> 4) * 8;         // A-matrix K base (layout split)
    const int px0  = wave * 32;               // first pixel of this wave's 2 M-tiles

    v8f acc00 = {}, acc01 = {}, acc02 = {}, acc03 = {};   // M-tile 0
    v8f acc10 = {}, acc11 = {}, acc12 = {}, acc13 = {};   // M-tile 1
    union U { v16h v; v8h h[2]; };

    for (int tap = 0; tap < 9; ++tap) {
        const int dy = tap / 3, dx = tap % 3;
        const _Float16* arow0 = sm + ((size_t)dy * LDSP + (px0 + m + dx)) * CCH;
        const _Float16* arow1 = arow0 + 16 * CCH;
#pragma unroll
        for (int q = 0; q < 2; ++q) {
            U a0, a1;
            a0.h[0] = *(const v8h*)(arow0 + q * 32 + kb);
            a0.h[1] = *(const v8h*)(arow0 + q * 32 + kb + 16);
            a1.h[0] = *(const v8h*)(arow1 + q * 32 + kb);
            a1.h[1] = *(const v8h*)(arow1 + q * 32 + kb + 16);
            const _Float16* wb = wp + ((size_t)((tap * 2 + q) * 4) * 32 + lane) * 16;
            U b0, b1, b2, b3;
            b0.h[0] = *(const v8h*)(wb +    0); b0.h[1] = *(const v8h*)(wb +    8);
            b1.h[0] = *(const v8h*)(wb +  512); b1.h[1] = *(const v8h*)(wb +  520);
            b2.h[0] = *(const v8h*)(wb + 1024); b2.h[1] = *(const v8h*)(wb + 1032);
            b3.h[0] = *(const v8h*)(wb + 1536); b3.h[1] = *(const v8h*)(wb + 1544);
            acc00 = __builtin_amdgcn_wmma_f32_16x16x32_f16(false, a0.v, false, b0.v, (short)0, acc00, false, false);
            acc10 = __builtin_amdgcn_wmma_f32_16x16x32_f16(false, a1.v, false, b0.v, (short)0, acc10, false, false);
            acc01 = __builtin_amdgcn_wmma_f32_16x16x32_f16(false, a0.v, false, b1.v, (short)0, acc01, false, false);
            acc11 = __builtin_amdgcn_wmma_f32_16x16x32_f16(false, a1.v, false, b1.v, (short)0, acc11, false, false);
            acc02 = __builtin_amdgcn_wmma_f32_16x16x32_f16(false, a0.v, false, b2.v, (short)0, acc02, false, false);
            acc12 = __builtin_amdgcn_wmma_f32_16x16x32_f16(false, a1.v, false, b2.v, (short)0, acc12, false, false);
            acc03 = __builtin_amdgcn_wmma_f32_16x16x32_f16(false, a0.v, false, b3.v, (short)0, acc03, false, false);
            acc13 = __builtin_amdgcn_wmma_f32_16x16x32_f16(false, a1.v, false, b3.v, (short)0, acc13, false, false);
        }
    }

    // ---- store: D layout lane = N, VGPR v = M (+8 for lanes 16-31) ----
    _Float16* orow = out + ((size_t)r * WW + c0 + px0) * CCH;
    const int n     = lane & 15;
    const int mbase = (lane >> 4) * 8;
#pragma unroll
    for (int v = 0; v < 8; ++v) {
        const size_t pb0 = (size_t)(mbase + v) * CCH;
        const size_t pb1 = (size_t)(16 + mbase + v) * CCH;
        orow[pb0 +  0 + n] = (_Float16)acc00[v];
        orow[pb0 + 16 + n] = (_Float16)acc01[v];
        orow[pb0 + 32 + n] = (_Float16)acc02[v];
        orow[pb0 + 48 + n] = (_Float16)acc03[v];
        orow[pb1 +  0 + n] = (_Float16)acc10[v];
        orow[pb1 + 16 + n] = (_Float16)acc11[v];
        orow[pb1 + 32 + n] = (_Float16)acc12[v];
        orow[pb1 + 48 + n] = (_Float16)acc13[v];
    }
}

// ---------------------------------------------------------------------------
// 1x1 head conv + bias.  mode 0: cls (nout=4, sigmoid+max/argmax)
//                        mode 1: two-plane raw output
// ---------------------------------------------------------------------------
__global__ __launch_bounds__(256) void head1x1(const _Float16* __restrict__ act,
                                               const float* __restrict__ wh,
                                               const float* __restrict__ bh,
                                               int nout, int mode,
                                               float* __restrict__ out_probs,
                                               int* __restrict__ out_ids,
                                               float* __restrict__ out_p0,
                                               float* __restrict__ out_p1) {
    int pix = blockIdx.x * 256 + threadIdx.x;
    if (pix >= HWPIX) return;
    const _Float16* a = act + (size_t)pix * CCH;
    float s[4];
    for (int j = 0; j < nout; ++j) s[j] = bh[j];
    for (int ci = 0; ci < CCH; ++ci) {
        float av = (float)a[ci];
        for (int j = 0; j < nout; ++j) s[j] += av * wh[j * CCH + ci];
    }
    if (mode == 0) {
        float best = -1.0f; int arg = 0;
        for (int j = 0; j < nout; ++j) {
            float p = 1.0f / (1.0f + expf(-s[j]));
            if (p > best) { best = p; arg = j; }
        }
        out_probs[pix] = best;
        out_ids[pix]   = arg;
    } else {
        out_p0[pix] = s[0];
        out_p1[pix] = s[1];
    }
}

// ---------------------------------------------------------------------------
// 3x3 peak mask + box decode.  out: [H*W][7] floats (zeros where !mask)
// ---------------------------------------------------------------------------
__global__ __launch_bounds__(256) void decode(const float* __restrict__ probs,
                                              const int* __restrict__ ids,
                                              const float* __restrict__ offp,
                                              const float* __restrict__ sizep,
                                              const float* __restrict__ yawp,
                                              const float* __restrict__ px_min,
                                              const float* __restrict__ py_min,
                                              const float* __restrict__ px_res,
                                              const float* __restrict__ py_res,
                                              const float* __restrict__ pthr,
                                              float* __restrict__ out) {
    int pix = blockIdx.x * 256 + threadIdx.x;
    if (pix >= HWPIX) return;
    int r = pix / WW, cw = pix % WW;
    float p = probs[pix];
    float pooled = -INFINITY;
    for (int dy = -1; dy <= 1; ++dy)
        for (int dx = -1; dx <= 1; ++dx) {
            int rr = r + dy, cc = cw + dx;
            if ((unsigned)rr < HH && (unsigned)cc < WW)
                pooled = fmaxf(pooled, probs[rr * WW + cc]);
        }
    bool mask = (p == pooled) && (p > pthr[0]);

    float dxv = offp[pix],        dyv = offp[HWPIX + pix];
    float lw  = sizep[pix],       ll  = sizep[HWPIX + pix];
    float ct  = yawp[pix],        st  = yawp[HWPIX + pix];

    float cx  = dxv + (float)cw * px_res[0] * 4.0f + px_min[0];
    float cy  = dyv + (float)r  * py_res[0] * 4.0f + py_min[0];
    float l   = expf(ll);
    float wv  = expf(lw);
    float ya  = atan2f(st, ct) * 0.5f;

    float* o = out + (size_t)pix * 7;
    if (mask) {
        o[0] = (float)ids[pix]; o[1] = p; o[2] = cx; o[3] = cy;
        o[4] = l; o[5] = wv; o[6] = ya;
    } else {
        o[0] = 0.f; o[1] = 0.f; o[2] = 0.f; o[3] = 0.f;
        o[4] = 0.f; o[5] = 0.f; o[6] = 0.f;
    }
}

// ---------------------------------------------------------------------------
extern "C" void kernel_launch(void* const* d_in, const int* in_sizes, int n_in,
                              void* d_out, int out_size, void* d_ws, size_t ws_size,
                              hipStream_t stream) {
    (void)in_sizes; (void)n_in; (void)out_size; (void)ws_size;

    const float* feat = (const float*)d_in[0];

    // workspace layout (bytes)
    char* ws = (char*)d_ws;
    const size_t ACT_B = (size_t)HWPIX * CCH * 2;            // 32 MB fp16
    _Float16* F  = (_Float16*)(ws);
    _Float16* A1 = (_Float16*)(ws + ACT_B);
    _Float16* A2 = (_Float16*)(ws + 2 * ACT_B);
    _Float16* WP = (_Float16*)(ws + 3 * ACT_B);              // 8 * 73728 B
    char* post = ws + 3 * ACT_B + (size_t)8 * WP_HALVES * 2;
    float* probs = (float*)post;                              // 1 MB
    int*   ids   = (int*)(post + 1 * (size_t)HWPIX * 4);      // 1 MB
    float* offb  = (float*)(post + 2 * (size_t)HWPIX * 4);    // 2 MB (2 planes)
    float* sizeb = (float*)(post + 4 * (size_t)HWPIX * 4);    // 2 MB
    float* yawb  = (float*)(post + 6 * (size_t)HWPIX * 4);    // 2 MB

    // 1) fp32 NCHW -> fp16 NHWC
    pack_feat<<<(CCH * HWPIX) / 256, 256, 0, stream>>>(feat, F);

    // 2) pack all 8 conv weights into B-fragment layout
    const int wsrc[8] = {1, 2, 5, 6, 9, 10, 13, 14};          // {w1,w2} x {cls,off,size,yaw}
    for (int k = 0; k < 8; ++k)
        pack_w<<<(WP_HALVES + 255) / 256, 256, 0, stream>>>(
            (const float*)d_in[wsrc[k]], WP + (size_t)k * WP_HALVES);

    // 3) per-head: conv -> conv -> 1x1 head
    dim3 cgrid(WW / TILE_W, HH);
    const int whsrc[4] = {3, 7, 11, 15};
    const int bhsrc[4] = {4, 8, 12, 16};
    for (int h = 0; h < 4; ++h) {
        conv3x3_wmma<<<cgrid, CONV_THREADS, 0, stream>>>(
            F,  WP + (size_t)(h * 2 + 0) * WP_HALVES, A1);
        conv3x3_wmma<<<cgrid, CONV_THREADS, 0, stream>>>(
            A1, WP + (size_t)(h * 2 + 1) * WP_HALVES, A2);
        const float* wh = (const float*)d_in[whsrc[h]];
        const float* bh = (const float*)d_in[bhsrc[h]];
        if (h == 0)
            head1x1<<<HWPIX / 256, 256, 0, stream>>>(A2, wh, bh, 4, 0, probs, ids, nullptr, nullptr);
        else {
            float* base = (h == 1) ? offb : (h == 2) ? sizeb : yawb;
            head1x1<<<HWPIX / 256, 256, 0, stream>>>(A2, wh, bh, 2, 1, nullptr, nullptr,
                                                     base, base + HWPIX);
        }
    }

    // 4) peak mask + box decode
    decode<<<HWPIX / 256, 256, 0, stream>>>(probs, ids, offb, sizeb, yawb,
                                            (const float*)d_in[17], (const float*)d_in[18],
                                            (const float*)d_in[19], (const float*)d_in[20],
                                            (const float*)d_in[21], (float*)d_out);
}
